// Geoattn_GNN_62156766708009
// MI455X (gfx1250) — compile-verified, matrix operand order checked by reading, not run on hardware
//
#include <hip/hip_runtime.h>

// ---------------------------------------------------------------------------
// Problem constants (from the reference)
// ---------------------------------------------------------------------------
static constexpr int NN  = 10000;    // nodes
static constexpr int NE  = 160000;   // edges
static constexpr int SCA = 256;
static constexpr int VEC = 64;
static constexpr int HID = 128;
#define GAUSS_STEP  (10.0f / 59.0f)
#define GAUSS_COEFF (-17.405f)       // -0.5/step^2 = -0.5*59^2/100

typedef __attribute__((ext_vector_type(16))) __bf16 v16bf;
typedef __attribute__((ext_vector_type(8)))  float  v8f;

// gfx1250 async global->LDS path (guarded; falls back to b128 load/store)
#if defined(__gfx1250__) && __has_builtin(__builtin_amdgcn_global_load_async_to_lds_b128)
#define USE_ASYNC_LDS 1
#else
#define USE_ASYNC_LDS 0
#endif

__device__ __forceinline__ void async_stage_wait() {
#if USE_ASYNC_LDS
#if __has_builtin(__builtin_amdgcn_s_wait_asynccnt)
  __builtin_amdgcn_s_wait_asynccnt(0);
#else
  asm volatile("s_wait_asynccnt 0x0" ::: "memory");
#endif
#endif
}

__device__ __forceinline__ void stage16B(const void* gsrc, void* ldst) {
#if USE_ASYNC_LDS
  // builtin signature (from clang diagnostic): arg0 = AS(1) int4-vector*,
  // arg1 = LDS int4-vector*, then imm offset, imm cpol
  typedef int v4i_t __attribute__((__vector_size__(16)));
  __builtin_amdgcn_global_load_async_to_lds_b128(
      (__attribute__((address_space(1))) v4i_t*)gsrc,
      (__attribute__((address_space(3))) v4i_t*)ldst, 0, 0);
#else
  *(uint4*)ldst = *(const uint4*)gsrc;
#endif
}

// ---------------------------------------------------------------------------
// small device helpers
// ---------------------------------------------------------------------------
__device__ __forceinline__ unsigned short f2bf(float f) {
  unsigned u = __float_as_uint(f);
  u += 0x7fffu + ((u >> 16) & 1u);           // round-to-nearest-even
  return (unsigned short)(u >> 16);
}
// monotone float<->uint mapping for atomicMax over signed floats
__device__ __forceinline__ unsigned encf(float f) {
  unsigned u = __float_as_uint(f);
  return (u & 0x80000000u) ? ~u : (u | 0x80000000u);
}
__device__ __forceinline__ float decf(unsigned k) {
  return (k & 0x80000000u) ? __uint_as_float(k & 0x7fffffffu)
                           : __uint_as_float(~k);
}

// ---------------------------------------------------------------------------
// Generic bf16 WMMA GEMM:  C[rows,Nout] = epi( A[rows,K] @ W[Nout,K]^T )
//   * A bf16, optionally row-gathered: src_row = idx[g/gd]*gd + g%gd
//     (idx1/segK gives a two-segment gather for the attention GEMM)
//   * Wb bf16 [Nout,K] packed (pre-converted once per launch)
//   * block = 128 threads (4 waves); tile 64x64x32; double-buffered LDS;
//     A/B staged with GLOBAL_LOAD_ASYNC_TO_LDS_B128 when available
// ---------------------------------------------------------------------------
struct GemmEpi {
  const float* bias;       // [Nout]
  int          act;        // 0 none, 1 sigmoid
  const float* colscale;   // [Nout]  (post-act multiply)
  const float* gate;       // [rows/gdiv, ldg] multiply
  int          gdiv, ldg;
  const float* r1row;      // rank-1: += r1row[row] * r1col[col]
  const float* r1col;
  float*          of;      // f32 out  [rows, Nout] (optional)
  unsigned short* ob;      // bf16 out [rows, Nout] (optional)
};

__global__ __launch_bounds__(128) void gemm_bf16_wmma(
    const unsigned short* __restrict__ A, int lda,
    const int* __restrict__ idx0, const int* __restrict__ idx1, int gd, int segK,
    const unsigned short* __restrict__ Wb,
    int rows, int K, int Nout, GemmEpi epi)
{
  __shared__ unsigned short As[2][64][40];   // 80B row pitch, 16B aligned slots
  __shared__ unsigned short Bs[2][64][40];
  const int rowBase = blockIdx.x * 64;
  const int colBase = blockIdx.y * 64;
  const int tid  = threadIdx.x;
  const int lane = tid & 31;
  const int wv   = tid >> 5;          // wave id 0..3 -> M subtile
  const int lr   = tid >> 1;          // staging row 0..63
  const int lh   = (tid & 1) * 16;    // staging col 0 or 16 (16 bf16 = 32B)

  // ---- hoist gather: resolve source row pointers once ----
  const int gA = rowBase + lr;
  const bool aValid = gA < rows;
  const unsigned short* aRow0 = nullptr;
  const unsigned short* aRow1 = nullptr;
  if (aValid) {
    long s0 = idx0 ? ((long)idx0[gA / gd] * gd + (gA % gd)) : (long)gA;
    aRow0 = A + s0 * (long)lda;
    if (idx1) aRow1 = A + ((long)idx1[gA / gd] * gd + (gA % gd)) * (long)lda;
  }
  const int gB = colBase + lr;
  const bool bValid = gB < Nout;
  const unsigned short* wRow = bValid ? (Wb + (long)gB * K) : nullptr;

  // ---- zero-fill tail rows once (never re-staged) ----
  {
    uint4 z = {0, 0, 0, 0};
    if (!aValid) {
      #pragma unroll
      for (int b = 0; b < 2; ++b) {
        *(uint4*)&As[b][lr][lh] = z; *(uint4*)&As[b][lr][lh + 8] = z;
      }
    }
    if (!bValid) {
      #pragma unroll
      for (int b = 0; b < 2; ++b) {
        *(uint4*)&Bs[b][lr][lh] = z; *(uint4*)&Bs[b][lr][lh + 8] = z;
      }
    }
  }

  auto stage = [&](int buf, int k0) {
    if (aValid) {
      const unsigned short* ar = aRow0;
      int kk = k0;
      if (idx1 && k0 >= segK) { ar = aRow1; kk = k0 - segK; }
      stage16B(ar + kk + lh,     &As[buf][lr][lh]);
      stage16B(ar + kk + lh + 8, &As[buf][lr][lh + 8]);
    }
    if (bValid) {
      stage16B(wRow + k0 + lh,     &Bs[buf][lr][lh]);
      stage16B(wRow + k0 + lh + 8, &Bs[buf][lr][lh + 8]);
    }
  };

  v8f acc[4];
  {
    v8f z = {0.f, 0.f, 0.f, 0.f, 0.f, 0.f, 0.f, 0.f};
    #pragma unroll
    for (int i = 0; i < 4; ++i) acc[i] = z;
  }

  const int nk = K >> 5;               // all K are multiples of 32
  stage(0, 0);
  const int m = lane & 15, hh = lane >> 4;
  for (int kt = 0; kt < nk; ++kt) {
    async_stage_wait();                // own async LDS writes landed
    __syncthreads();                   // everyone's tile visible
    if (kt + 1 < nk) stage((kt + 1) & 1, (kt + 1) << 5);
    const int buf = kt & 1;
    // ISA 7.12.2 16-bit A 16x32: lane m=l&15, half h=l>>4,
    //   vgpr i<4: K = 2i + 8h (pair); vgpr i>=4: K = 16 + 2(i-4) + 8h
    union { v16bf v; unsigned u[8]; } a, b;
    #pragma unroll
    for (int i = 0; i < 4; ++i) {
      a.u[i]     = *(const unsigned*)&As[buf][wv * 16 + m][2 * i + 8 * hh];
      a.u[i + 4] = *(const unsigned*)&As[buf][wv * 16 + m][16 + 2 * i + 8 * hh];
    }
    #pragma unroll
    for (int nt = 0; nt < 4; ++nt) {
      #pragma unroll
      for (int i = 0; i < 4; ++i) {
        b.u[i]     = *(const unsigned*)&Bs[buf][nt * 16 + m][2 * i + 8 * hh];
        b.u[i + 4] = *(const unsigned*)&Bs[buf][nt * 16 + m][16 + 2 * i + 8 * hh];
      }
      acc[nt] = __builtin_amdgcn_wmma_f32_16x16x32_bf16(
          false, a.v, false, b.v, (short)0, acc[nt], false, false);
    }
  }

  // epilogue: C layout — lane n=l&15, half h=l>>4; vgpr r -> row r + 8h
  #pragma unroll
  for (int nt = 0; nt < 4; ++nt) {
    int gcol = colBase + nt * 16 + m;
    if (gcol >= Nout) continue;
    #pragma unroll
    for (int r = 0; r < 8; ++r) {
      int grow = rowBase + wv * 16 + hh * 8 + r;
      if (grow >= rows) continue;
      float v = acc[nt][r];
      if (epi.r1row)    v += epi.r1row[grow] * epi.r1col[gcol];
      if (epi.bias)     v += epi.bias[gcol];
      if (epi.act == 1) v = 1.f / (1.f + __expf(-v));
      if (epi.colscale) v *= epi.colscale[gcol];
      if (epi.gate)     v *= epi.gate[(long)(grow / epi.gdiv) * epi.ldg + gcol];
      long o = (long)grow * Nout + gcol;
      if (epi.of) epi.of[o] = v;
      if (epi.ob) epi.ob[o] = f2bf(v);
    }
  }
}

// ---------------------------------------------------------------------------
// elementwise / utility kernels
// ---------------------------------------------------------------------------
__global__ void k_fill_u32(unsigned* p, unsigned v, long n) {
  long g = (long)blockIdx.x * blockDim.x + threadIdx.x;
  if (g < n) p[g] = v;
}
__global__ void k_fill_f32(float* p, float v, long n) {
  long g = (long)blockIdx.x * blockDim.x + threadIdx.x;
  if (g < n) p[g] = v;
}
// pack a f32 [rowsW, ldin] weight into bf16 [rowsW, Kc]
__global__ void k_w2bf(const float* in, int ldin, unsigned short* out, long n, int Kc) {
  long g = (long)blockIdx.x * blockDim.x + threadIdx.x;
  if (g >= n) return;
  long r = g / Kc; int k = (int)(g % Kc);
  out[g] = f2bf(in[r * (long)ldin + k]);
}

// u = em_Wv1 @ W_em[:,0];  uabs=|u|;  w2u = em_Wv2 @ u   (rank-1 edge-vec basis)
__global__ void k_u(const float* Wv1, const float* Wv2, const float* Wem,
                    float* uabs, float* w2u) {
  __shared__ float u[64];
  int c = threadIdx.x;
  float s = 0.f;
  for (int i = 0; i < 64; ++i) s += Wv1[c * 64 + i] * Wem[i];
  u[c] = s; uabs[c] = fabsf(s);
  __syncthreads();
  float t = 0.f;
  for (int h = 0; h < 64; ++h) t += Wv2[c * 64 + h] * u[h];
  w2u[c] = t;
}
__global__ void k_wlast(const float* Wattn, float* wlast) {
  int n = threadIdx.x;            // 128
  wlast[n] = Wattn[(long)n * 513 + 512];
}
__global__ void k_row2(const int* row, const int* col, int* row2, int* col2, int E) {
  int e = blockIdx.x * blockDim.x + threadIdx.x;
  if (e >= E) return;
  row2[e] = row[row[e]];
  col2[e] = row[col[e]];
}

// per-edge: dist, normalized ev, em-mapper Ws input [ |u|*|ev| (64) | gauss+feat (64) ]
__global__ void k_edge_feat(const float* evec, const float* efeat, const float* uabs,
                            float* ev, float* dist, unsigned short* Aem, int E) {
  long gid = (long)blockIdx.x * blockDim.x + threadIdx.x;
  if (gid >= (long)E * 128) return;
  int e = (int)(gid >> 7), c = (int)(gid & 127);
  float x = evec[e * 3 + 0], y = evec[e * 3 + 1], z = evec[e * 3 + 2];
  float d = sqrtf(x * x + y * y + z * z);
  float inv = 1.f / (d + 1e-7f);
  if (c == 0) { dist[e] = d; ev[e*3+0] = x*inv; ev[e*3+1] = y*inv; ev[e*3+2] = z*inv; }
  float val;
  if (c < 64) {
    val = uabs[c] * (d * inv);               // |v_inter| = |u[c]| * |ev|
  } else {
    int cc = c - 64;
    if (cc < 60) { float t = d - GAUSS_STEP * (float)cc; val = __expf(GAUSS_COEFF * t * t); }
    else val = efeat[(long)e * 4 + (cc - 60)];
  }
  Aem[gid] = f2bf(val);
}

// [R,C,3] f32 -> [R,3,C] bf16  (x-major layout for VN GEMMs)
__global__ void k_vec_to_xmajor(const float* in, unsigned short* out, long R, int C) {
  long gid = (long)blockIdx.x * blockDim.x + threadIdx.x;
  if (gid >= R * 3 * C) return;
  int c = (int)(gid % C); long t = gid / C;
  int x = (int)(t % 3);   long n = t / 3;
  out[gid] = f2bf(in[(n * (long)C + c) * 3 + x]);
}

// vnorm over x of vi3 [R,3,C] -> bf16 into cat[r, coloff + c] (ld = ldcat)
__global__ void k_norm3(const float* vi3, unsigned short* cat,
                        long R, int C, int ldcat, int coloff) {
  long gid = (long)blockIdx.x * blockDim.x + threadIdx.x;
  if (gid >= R * C) return;
  long r = gid / C; int c = (int)(gid % C);
  float a = vi3[(r * 3 + 0) * (long)C + c];
  float b = vi3[(r * 3 + 1) * (long)C + c];
  float d = vi3[(r * 3 + 2) * (long)C + c];
  cat[r * (long)ldcat + coloff + c] = f2bf(sqrtf(a*a + b*b + d*d));
}
// f32 [R,ldin] -> bf16 into cat[r, coloff + c]
__global__ void k_cat_copy(const float* in, int ldin, unsigned short* cat,
                           long R, int C, int ldcat, int coloff) {
  long gid = (long)blockIdx.x * blockDim.x + threadIdx.x;
  if (gid >= R * C) return;
  long r = gid / C; int c = (int)(gid % C);
  cat[r * (long)ldcat + coloff + c] = f2bf(in[r * (long)ldin + c]);
}

// alpha_vec[e] = sigmoid( <avh3[row[e]], avh3[col[e]]> ), 384-dot, 1 wave/edge
__global__ void k_alpha_vec(const float* avh3, const int* row, const int* col,
                            float* av, int E) {
  int lane = threadIdx.x & 31;
  int e = blockIdx.x * (blockDim.x >> 5) + (threadIdx.x >> 5);
  if (e >= E) return;
  const float* pa = avh3 + (long)row[e] * 384;
  const float* pb = avh3 + (long)col[e] * 384;
  float s = 0.f;
  for (int i = lane; i < 384; i += 32) s += pa[i] * pb[i];
  for (int o = 16; o > 0; o >>= 1) s += __shfl_down(s, o, 32);
  if (lane == 0) av[e] = 1.f / (1.f + __expf(-s));
}

// scatter-softmax over segments `row`
__global__ void k_smax1(const float* a, const int* row, unsigned* mx, int E) {
  long gid = (long)blockIdx.x * blockDim.x + threadIdx.x;
  if (gid >= (long)E * HID) return;
  int e = (int)(gid >> 7), h = (int)(gid & 127);
  atomicMax(&mx[(long)row[e] * HID + h], encf(a[gid]));
}
__global__ void k_smax2(float* a, const int* row, const unsigned* mx, float* ssum, int E) {
  long gid = (long)blockIdx.x * blockDim.x + threadIdx.x;
  if (gid >= (long)E * HID) return;
  int e = (int)(gid >> 7), h = (int)(gid & 127);
  float ex = __expf(a[gid] - decf(mx[(long)row[e] * HID + h]));
  a[gid] = ex;
  atomicAdd(&ssum[(long)row[e] * HID + h], ex);
}
__global__ void k_smax3(float* a, const int* row, const float* ssum, int E) {
  long gid = (long)blockIdx.x * blockDim.x + threadIdx.x;
  if (gid >= (long)E * HID) return;
  int e = (int)(gid >> 7), h = (int)(gid & 127);
  a[gid] = a[gid] / (ssum[(long)row[e] * HID + h] + 1e-16f);
}

// assemble messages: msg_sca -> cat_mo[:,128:256]; msg_vec -> bf16 [E,3,128]
__global__ void k_msg_build(const int* row2, const float* nn_out, const float* en_out,
                            const float* alpha, const float* nsn_out, const float* esn_out,
                            const float* evh, const float* ev, const float* nvn3,
                            const float* alpha_vec, unsigned short* cat_mo,
                            unsigned short* msg_vec3b, int E) {
  long gid = (long)blockIdx.x * blockDim.x + threadIdx.x;
  if (gid >= (long)E * HID) return;
  int e = (int)(gid >> 7), h = (int)(gid & 127);
  long r2 = row2[e];
  float al = alpha[gid];
  float ms = nn_out[r2 * HID + h] * en_out[gid] * al;
  cat_mo[(long)e * 256 + 128 + h] = f2bf(ms);
  float nsn = nsn_out[r2 * HID + h];
  float esn = esn_out[gid];
  float ehc = evh[gid];
  float avv = alpha_vec[e];
  #pragma unroll
  for (int x = 0; x < 3; ++x) {
    float mv = (nsn * ehc * ev[(long)e * 3 + x] +
                nvn3[(r2 * 3 + x) * HID + h] * esn) * avv;
    msg_vec3b[((long)e * 3 + x) * HID + h] = f2bf(mv);
  }
}

__global__ void k_scatter_sca(const float* s, const int* row, float* aggr, int E) {
  long gid = (long)blockIdx.x * blockDim.x + threadIdx.x;
  if (gid >= (long)E * SCA) return;
  int e = (int)(gid >> 8), c = (int)(gid & 255);
  atomicAdd(&aggr[(long)row[e] * SCA + c], s[gid]);
}
__global__ void k_scatter_vec(const float* v, const int* row, float* aggr, int E) {
  long gid = (long)blockIdx.x * blockDim.x + threadIdx.x;
  if (gid >= (long)E * 192) return;
  int e = (int)(gid / 192), r = (int)(gid % 192);
  atomicAdd(&aggr[(long)row[e] * 192 + r], v[gid]);
}
__global__ void k_add_sca(const float* a, const float* b, float* o, long n) {
  long g = (long)blockIdx.x * blockDim.x + threadIdx.x;
  if (g < n) o[g] = a[g] + b[g];
}
__global__ void k_add_vec_bf16(const float* a, const float* b, unsigned short* o, long n) {
  long g = (long)blockIdx.x * blockDim.x + threadIdx.x;
  if (g < n) o[g] = f2bf(a[g] + b[g]);
}

// out_sca leaky(0.01)
__global__ void k_final_sca(const float* s, float* out, long n) {
  long g = (long)blockIdx.x * blockDim.x + threadIdx.x;
  if (g >= n) return;
  float v = s[g];
  out[g] = v >= 0.f ? v : 0.01f * v;
}
// VN-LeakyReLU + transpose back to [N,64,3]
__global__ void k_final_vec(const float* vec3, const float* d3, float* out, int N) {
  long gid = (long)blockIdx.x * blockDim.x + threadIdx.x;
  if (gid >= (long)N * VEC) return;
  long n = gid / VEC; int c = (int)(gid % VEC);
  float v[3], dd[3];
  #pragma unroll
  for (int x = 0; x < 3; ++x) {
    v[x]  = vec3[(n * 3 + x) * VEC + c];
    dd[x] = d3[(n * 3 + x) * VEC + c];
  }
  float dot = v[0]*dd[0] + v[1]*dd[1] + v[2]*dd[2];
  float dns = dd[0]*dd[0] + dd[1]*dd[1] + dd[2]*dd[2];
  float f = dot / (dns + 1e-6f);
  #pragma unroll
  for (int x = 0; x < 3; ++x) {
    float pj = v[x] - f * dd[x];
    float w  = (dot >= 0.f) ? v[x] : pj;
    out[n * 192 + c * 3 + x] = 0.2f * v[x] + 0.8f * w;
  }
}

// ---------------------------------------------------------------------------
// host side
// ---------------------------------------------------------------------------
static GemmEpi epi0() {
  GemmEpi e; e.bias = nullptr; e.act = 0; e.colscale = nullptr; e.gate = nullptr;
  e.gdiv = 1; e.ldg = 0; e.r1row = nullptr; e.r1col = nullptr;
  e.of = nullptr; e.ob = nullptr; return e;
}
static inline long cdivl(long a, long b) { return (a + b - 1) / b; }

static void gemm(hipStream_t s, const unsigned short* A, int lda,
                 const unsigned short* Wb, int rows, int K, int Nout, GemmEpi epi,
                 const int* idx0 = nullptr, const int* idx1 = nullptr,
                 int gd = 1, int segK = 1 << 30) {
  dim3 grid((unsigned)cdivl(rows, 64), (unsigned)cdivl(Nout, 64));
  gemm_bf16_wmma<<<grid, 128, 0, s>>>(A, lda, idx0, idx1, gd, segK,
                                      Wb, rows, K, Nout, epi);
}

extern "C" void kernel_launch(void* const* d_in, const int* in_sizes, int n_in,
                              void* d_out, int out_size, void* d_ws, size_t ws_size,
                              hipStream_t stream) {
  // ---- inputs (setup_inputs dict order; params dict in insertion order) ----
  const float* node_sca = (const float*)d_in[0];
  const float* node_vec = (const float*)d_in[1];
  const float* edge_feat = (const float*)d_in[2];
  const float* edge_vector = (const float*)d_in[3];
  const int*   row = (const int*)d_in[4];
  const int*   col = row + NE;
  int pi = 5;
  const float* W_em = (const float*)d_in[pi++];
  struct GV { const float *Wv1, *Wv2, *Ws, *Wg, *bg; } nm, em, mo, rc, ao;
  GV* gvs[5] = {&nm, &em, &mo, &rc, &ao};
  for (int i = 0; i < 5; ++i) {
    gvs[i]->Wv1 = (const float*)d_in[pi++];
    gvs[i]->Wv2 = (const float*)d_in[pi++];
    gvs[i]->Ws  = (const float*)d_in[pi++];
    gvs[i]->Wg  = (const float*)d_in[pi++];
    gvs[i]->bg  = (const float*)d_in[pi++];
  }
  const float *W_en = (const float*)d_in[pi++], *b_en = (const float*)d_in[pi++];
  const float *W_nn = (const float*)d_in[pi++], *b_nn = (const float*)d_in[pi++];
  const float *W_esn = (const float*)d_in[pi++], *b_esn = (const float*)d_in[pi++];
  const float *W_nsn = (const float*)d_in[pi++], *b_nsn = (const float*)d_in[pi++];
  const float *W_attn = (const float*)d_in[pi++], *b_attn = (const float*)d_in[pi++];
  const float *W_evn = (const float*)d_in[pi++];
  const float *W_nvn = (const float*)d_in[pi++];
  const float *W_vattn = (const float*)d_in[pi++];
  const float *ao_Wdir = (const float*)d_in[pi++];

  float* out_sca = (float*)d_out;
  float* out_vec = out_sca + (long)NN * SCA;

  // ---- workspace (bump allocator; two explicit alias groups) ----
  char* ws = (char*)d_ws;
  size_t off = 0;
  auto alloc = [&](size_t bytes) -> void* {
    void* p = ws + off; off = (off + bytes + 255) & ~(size_t)255; return p;
  };
  // alias group X: en|esn|alpha later reused as vi3_mo f32 [3E,128]
  float* en_out  = (float*)alloc((size_t)NE * HID * 4);
  float* esn_out = (float*)alloc((size_t)NE * HID * 4);
  float* alpha   = (float*)alloc((size_t)NE * HID * 4);
  float* vi3_mo  = en_out;                                   // 3E*128 f32
  // alias group Y: A_em|Sb_em later reused as vout3_mo f32 [3E,64]
  unsigned short* A_em  = (unsigned short*)alloc((size_t)NE * 128 * 2);
  unsigned short* Sb_em = (unsigned short*)alloc((size_t)NE * 256 * 2);
  float* vout3_mo = (float*)A_em;                            // 3E*64 f32
  // edge buffers
  unsigned short* evec_coef = (unsigned short*)alloc((size_t)NE * 64 * 2);
  float* evh_coef = (float*)alloc((size_t)NE * HID * 4);
  float* ev   = (float*)alloc((size_t)NE * 3 * 4);
  float* dist = (float*)alloc((size_t)NE * 4);
  int* row2 = (int*)alloc((size_t)NE * 4);
  int* col2 = (int*)alloc((size_t)NE * 4);
  float* alpha_vec = (float*)alloc((size_t)NE * 4);
  unsigned short* cat_mo    = (unsigned short*)alloc((size_t)NE * 256 * 2);
  unsigned short* msg_vec3b = (unsigned short*)alloc((size_t)NE * 3 * HID * 2);
  unsigned short* vi3b_mo   = (unsigned short*)alloc((size_t)NE * 3 * HID * 2);
  float* S_mo = (float*)alloc((size_t)NE * 256 * 4);
  unsigned short* Sb_mo = (unsigned short*)alloc((size_t)NE * 256 * 2);
  float* gate_mo = (float*)alloc((size_t)NE * 64 * 4);
  // node buffers
  unsigned short* nvin3b = (unsigned short*)alloc((size_t)NN * 192 * 2);
  float* vi3_nm = (float*)alloc((size_t)NN * 192 * 4);
  unsigned short* vi3b_nm = (unsigned short*)alloc((size_t)NN * 192 * 2);
  unsigned short* cat_nm = (unsigned short*)alloc((size_t)NN * 320 * 2);
  unsigned short* nsf_bf = (unsigned short*)alloc((size_t)NN * 256 * 2);
  float* gate_nm = (float*)alloc((size_t)NN * 64 * 4);
  unsigned short* nvf3b = (unsigned short*)alloc((size_t)NN * 192 * 2);
  float* nn_out  = (float*)alloc((size_t)NN * HID * 4);
  float* nsn_out = (float*)alloc((size_t)NN * HID * 4);
  float* nvn3 = (float*)alloc((size_t)NN * 3 * HID * 4);
  float* avh3 = (float*)alloc((size_t)NN * 3 * HID * 4);
  float* vi3_rc = (float*)alloc((size_t)NN * 192 * 4);
  unsigned short* vi3b_rc = (unsigned short*)alloc((size_t)NN * 192 * 2);
  unsigned short* cat_rc = (unsigned short*)alloc((size_t)NN * 320 * 2);
  float* S_rc = (float*)alloc((size_t)NN * 256 * 4);
  unsigned short* Sb_rc = (unsigned short*)alloc((size_t)NN * 256 * 2);
  float* gate_rc = (float*)alloc((size_t)NN * 64 * 4);
  float* vout3_rc = (float*)alloc((size_t)NN * 192 * 4);
  unsigned* mx = (unsigned*)alloc((size_t)NN * HID * 4);
  float* ssum = (float*)alloc((size_t)NN * HID * 4);
  float* aggr_sca  = (float*)alloc((size_t)NN * 256 * 4);
  float* aggr_vec3 = (float*)alloc((size_t)NN * 192 * 4);
  float* in_sca = (float*)alloc((size_t)NN * 256 * 4);
  unsigned short* invec3b = (unsigned short*)alloc((size_t)NN * 192 * 2);
  float* vi3_ao = (float*)alloc((size_t)NN * 192 * 4);
  unsigned short* vi3b_ao = (unsigned short*)alloc((size_t)NN * 192 * 2);
  unsigned short* cat_ao = (unsigned short*)alloc((size_t)NN * 320 * 2);
  float* S_ao = (float*)alloc((size_t)NN * 256 * 4);
  unsigned short* Sb_ao = (unsigned short*)alloc((size_t)NN * 256 * 2);
  float* gate_ao = (float*)alloc((size_t)NN * 64 * 4);
  float* vout3_ao = (float*)alloc((size_t)NN * 192 * 4);
  unsigned short* vout3b_ao = (unsigned short*)alloc((size_t)NN * 192 * 2);
  float* d3 = (float*)alloc((size_t)NN * 192 * 4);
  float* uabs = (float*)alloc(64 * 4);
  float* w2u  = (float*)alloc(64 * 4);
  float* wlast = (float*)alloc(128 * 4);
  (void)ws_size; (void)n_in; (void)in_sizes; (void)out_size;

  auto ew = [&](long n) { return dim3((unsigned)cdivl(n, 256)); };

  // ---- pre-pack all GEMM weights to bf16 once per launch ----
  auto wconv = [&](const float* W, int rowsW, int Kc, int ldin) -> const unsigned short* {
    unsigned short* o = (unsigned short*)alloc((size_t)rowsW * Kc * 2);
    long n = (long)rowsW * Kc;
    k_w2bf<<<ew(n), 256, 0, stream>>>(W, ldin, o, n, Kc);
    return o;
  };
  const unsigned short* nmWv1b = wconv(nm.Wv1, 64, 64, 64);
  const unsigned short* nmWsb  = wconv(nm.Ws, 256, 320, 320);
  const unsigned short* nmWgb  = wconv(nm.Wg, 64, 256, 256);
  const unsigned short* nmWv2b = wconv(nm.Wv2, 64, 64, 64);
  const unsigned short* rcWv1b = wconv(rc.Wv1, 64, 64, 64);
  const unsigned short* rcWsb  = wconv(rc.Ws, 256, 320, 320);
  const unsigned short* rcWgb  = wconv(rc.Wg, 64, 256, 256);
  const unsigned short* rcWv2b = wconv(rc.Wv2, 64, 64, 64);
  const unsigned short* aoWv1b = wconv(ao.Wv1, 64, 64, 64);
  const unsigned short* aoWsb  = wconv(ao.Ws, 256, 320, 320);
  const unsigned short* aoWgb  = wconv(ao.Wg, 64, 256, 256);
  const unsigned short* aoWv2b = wconv(ao.Wv2, 64, 64, 64);
  const unsigned short* emWsb  = wconv(em.Ws, 256, 128, 128);
  const unsigned short* emWgb  = wconv(em.Wg, 64, 256, 256);
  const unsigned short* moWv1b = wconv(mo.Wv1, 128, 128, 128);
  const unsigned short* moWsb  = wconv(mo.Ws, 256, 256, 256);
  const unsigned short* moWgb  = wconv(mo.Wg, 64, 256, 256);
  const unsigned short* moWv2b = wconv(mo.Wv2, 64, 128, 128);
  const unsigned short* Wnnb   = wconv(W_nn, 128, 256, 256);
  const unsigned short* Wnsnb  = wconv(W_nsn, 128, 256, 256);
  const unsigned short* Wenb   = wconv(W_en, 128, 256, 256);
  const unsigned short* Wesnb  = wconv(W_esn, 128, 256, 256);
  const unsigned short* Wnvnb  = wconv(W_nvn, 128, 64, 64);
  const unsigned short* Wvatb  = wconv(W_vattn, 128, 64, 64);
  const unsigned short* Wevnb  = wconv(W_evn, 128, 64, 64);
  const unsigned short* Wattnb = wconv(W_attn, 128, 512, 513);   // drop col 512
  const unsigned short* Wdirb  = wconv(ao_Wdir, 64, 64, 64);

  // ---- tiny precomputes ----
  k_u<<<1, 64, 0, stream>>>(em.Wv1, em.Wv2, W_em, uabs, w2u);
  k_wlast<<<1, 128, 0, stream>>>(W_attn, wlast);
  k_row2<<<ew(NE), 256, 0, stream>>>(row, col, row2, col2, NE);

  // ---- node_mapper (nm) + per-node projections ----
  k_vec_to_xmajor<<<ew((long)NN * 192), 256, 0, stream>>>(node_vec, nvin3b, NN, VEC);
  { GemmEpi e = epi0(); e.of = vi3_nm; e.ob = vi3b_nm;
    gemm(stream, nvin3b, 64, nmWv1b, 3 * NN, 64, 64, e); }
  k_norm3<<<ew((long)NN * 64), 256, 0, stream>>>(vi3_nm, cat_nm, NN, 64, 320, 0);
  k_cat_copy<<<ew((long)NN * 256), 256, 0, stream>>>(node_sca, 256, cat_nm, NN, 256, 320, 64);
  { GemmEpi e = epi0(); e.ob = nsf_bf;
    gemm(stream, cat_nm, 320, nmWsb, NN, 320, 256, e); }
  { GemmEpi e = epi0(); e.bias = nm.bg; e.act = 1; e.of = gate_nm;
    gemm(stream, nsf_bf, 256, nmWgb, NN, 256, 64, e); }
  { GemmEpi e = epi0(); e.gate = gate_nm; e.gdiv = 3; e.ldg = 64; e.ob = nvf3b;
    gemm(stream, vi3b_nm, 64, nmWv2b, 3 * NN, 64, 64, e); }
  { GemmEpi e = epi0(); e.bias = b_nn; e.of = nn_out;
    gemm(stream, nsf_bf, 256, Wnnb, NN, 256, HID, e); }
  { GemmEpi e = epi0(); e.bias = b_nsn; e.of = nsn_out;
    gemm(stream, nsf_bf, 256, Wnsnb, NN, 256, HID, e); }
  { GemmEpi e = epi0(); e.of = nvn3;
    gemm(stream, nvf3b, 64, Wnvnb, 3 * NN, 64, HID, e); }
  { GemmEpi e = epi0(); e.of = avh3;        // avh[j] = W_vattn @ nvf[row[j]]
    gemm(stream, nvf3b, 64, Wvatb, 3 * NN, 64, HID, e, row, nullptr, 3); }

  // ---- resi_connecter (rc) ----
  { GemmEpi e = epi0(); e.of = vi3_rc; e.ob = vi3b_rc;
    gemm(stream, nvin3b, 64, rcWv1b, 3 * NN, 64, 64, e); }
  k_norm3<<<ew((long)NN * 64), 256, 0, stream>>>(vi3_rc, cat_rc, NN, 64, 320, 0);
  k_cat_copy<<<ew((long)NN * 256), 256, 0, stream>>>(node_sca, 256, cat_rc, NN, 256, 320, 64);
  { GemmEpi e = epi0(); e.of = S_rc; e.ob = Sb_rc;
    gemm(stream, cat_rc, 320, rcWsb, NN, 320, 256, e); }
  { GemmEpi e = epi0(); e.bias = rc.bg; e.act = 1; e.of = gate_rc;
    gemm(stream, Sb_rc, 256, rcWgb, NN, 256, 64, e); }
  { GemmEpi e = epi0(); e.gate = gate_rc; e.gdiv = 3; e.ldg = 64; e.of = vout3_rc;
    gemm(stream, vi3b_rc, 64, rcWv2b, 3 * NN, 64, 64, e); }

  // ---- edge featurization + edge_mapper (em, rank-1 vector path) ----
  k_edge_feat<<<ew((long)NE * 128), 256, 0, stream>>>(edge_vector, edge_feat, uabs,
                                                      ev, dist, A_em, NE);
  { GemmEpi e = epi0(); e.ob = Sb_em;
    gemm(stream, A_em, 128, emWsb, NE, 128, 256, e); }
  { GemmEpi e = epi0(); e.bias = em.bg; e.act = 1; e.colscale = w2u; e.ob = evec_coef;
    gemm(stream, Sb_em, 256, emWgb, NE, 256, 64, e); }
  { GemmEpi e = epi0(); e.of = evh_coef;
    gemm(stream, evec_coef, 64, Wevnb, NE, 64, HID, e); }
  { GemmEpi e = epi0(); e.bias = b_en; e.of = en_out;
    gemm(stream, Sb_em, 256, Wenb, NE, 256, HID, e); }
  { GemmEpi e = epi0(); e.bias = b_esn; e.of = esn_out;
    gemm(stream, Sb_em, 256, Wesnb, NE, 256, HID, e); }

  // ---- attention (two-segment gathered K=512 GEMM + rank-1 dist column) ----
  { GemmEpi e = epi0(); e.bias = b_attn; e.r1row = dist; e.r1col = wlast; e.of = alpha;
    gemm(stream, nsf_bf, 256, Wattnb, NE, 512, HID, e, row2, col2, 1, 256); }
  k_fill_u32<<<ew((long)NN * HID), 256, 0, stream>>>(mx, 0u, (long)NN * HID);
  k_fill_f32<<<ew((long)NN * HID), 256, 0, stream>>>(ssum, 0.f, (long)NN * HID);
  k_smax1<<<ew((long)NE * HID), 256, 0, stream>>>(alpha, row, mx, NE);
  k_smax2<<<ew((long)NE * HID), 256, 0, stream>>>(alpha, row, mx, ssum, NE);
  k_smax3<<<ew((long)NE * HID), 256, 0, stream>>>(alpha, row, ssum, NE);
  k_alpha_vec<<<dim3(NE / 8), 256, 0, stream>>>(avh3, row, col, alpha_vec, NE);

  // ---- message assembly + msg_out (mo) ----
  k_msg_build<<<ew((long)NE * HID), 256, 0, stream>>>(row2, nn_out, en_out, alpha,
      nsn_out, esn_out, evh_coef, ev, nvn3, alpha_vec, cat_mo, msg_vec3b, NE);
  { GemmEpi e = epi0(); e.of = vi3_mo; e.ob = vi3b_mo;     // vi3_mo aliases en/esn/alpha
    gemm(stream, msg_vec3b, 128, moWv1b, 3 * NE, 128, 128, e); }
  k_norm3<<<ew((long)NE * 128), 256, 0, stream>>>(vi3_mo, cat_mo, NE, 128, 256, 0);
  { GemmEpi e = epi0(); e.of = S_mo; e.ob = Sb_mo;
    gemm(stream, cat_mo, 256, moWsb, NE, 256, 256, e); }
  { GemmEpi e = epi0(); e.bias = mo.bg; e.act = 1; e.of = gate_mo;
    gemm(stream, Sb_mo, 256, moWgb, NE, 256, 64, e); }
  { GemmEpi e = epi0(); e.gate = gate_mo; e.gdiv = 3; e.ldg = 64; e.of = vout3_mo;
    gemm(stream, vi3b_mo, 128, moWv2b, 3 * NE, 128, 64, e); }

  // ---- aggregate + residual ----
  k_fill_f32<<<ew((long)NN * 256), 256, 0, stream>>>(aggr_sca, 0.f, (long)NN * 256);
  k_fill_f32<<<ew((long)NN * 192), 256, 0, stream>>>(aggr_vec3, 0.f, (long)NN * 192);
  k_scatter_sca<<<ew((long)NE * 256), 256, 0, stream>>>(S_mo, row, aggr_sca, NE);
  k_scatter_vec<<<ew((long)NE * 192), 256, 0, stream>>>(vout3_mo, row, aggr_vec3, NE);
  k_add_sca<<<ew((long)NN * 256), 256, 0, stream>>>(S_rc, aggr_sca, in_sca, (long)NN * 256);
  k_add_vec_bf16<<<ew((long)NN * 192), 256, 0, stream>>>(vout3_rc, aggr_vec3, invec3b,
                                                         (long)NN * 192);

  // ---- aggr_out (ao) + VN-LeakyReLU + output ----
  { GemmEpi e = epi0(); e.of = vi3_ao; e.ob = vi3b_ao;
    gemm(stream, invec3b, 64, aoWv1b, 3 * NN, 64, 64, e); }
  k_norm3<<<ew((long)NN * 64), 256, 0, stream>>>(vi3_ao, cat_ao, NN, 64, 320, 0);
  k_cat_copy<<<ew((long)NN * 256), 256, 0, stream>>>(in_sca, 256, cat_ao, NN, 256, 320, 64);
  { GemmEpi e = epi0(); e.of = S_ao; e.ob = Sb_ao;
    gemm(stream, cat_ao, 320, aoWsb, NN, 320, 256, e); }
  { GemmEpi e = epi0(); e.bias = ao.bg; e.act = 1; e.of = gate_ao;
    gemm(stream, Sb_ao, 256, aoWgb, NN, 256, 64, e); }
  { GemmEpi e = epi0(); e.gate = gate_ao; e.gdiv = 3; e.ldg = 64;
    e.of = vout3_ao; e.ob = vout3b_ao;
    gemm(stream, vi3b_ao, 64, aoWv2b, 3 * NN, 64, 64, e); }
  { GemmEpi e = epi0(); e.of = d3;
    gemm(stream, vout3b_ao, 64, Wdirb, 3 * NN, 64, 64, e); }
  k_final_sca<<<ew((long)NN * 256), 256, 0, stream>>>(S_ao, out_sca, (long)NN * 256);
  k_final_vec<<<ew((long)NN * 64), 256, 0, stream>>>(vout3_ao, d3, out_vec, NN);
}
